// Network_60163901882451
// MI455X (gfx1250) — compile-verified, matrix operand order checked by reading, not run on hardware
//
#include <hip/hip_runtime.h>
#include <hip/hip_bf16.h>

// ---------------- problem constants ----------------
#define BATCH   64
#define NODES   128
#define SAMP    (BATCH*NODES)      // 8192
#define P81     81                 // 9x9 spatial
#define C1      32
#define C2      64
#define C3      128
#define FLATK   10368              // 128*81
#define ENC     128
#define ACTS    5

#define AS1 __attribute__((address_space(1)))
#define AS3 __attribute__((address_space(3)))

// gfx1250 async global->LDS path (ASYNCcnt), with sync fallback
#if defined(__HIP_DEVICE_COMPILE__) && \
    __has_builtin(__builtin_amdgcn_global_load_async_to_lds_b128) && \
    __has_builtin(__builtin_amdgcn_s_wait_asynccnt)
#define HAVE_ASYNC_LDS 1
#endif

typedef __attribute__((ext_vector_type(16))) __bf16 v16bf;
typedef __attribute__((ext_vector_type(8)))  float  f32x8;

#if HAVE_ASYNC_LDS
typedef int v4i __attribute__((vector_size(16)));
// GLOBAL_LOAD_ASYNC_TO_LDS_B128: 16B per lane, ASYNCcnt-tracked
__device__ __forceinline__ void async_ld16(unsigned short* lds_dst,
                                           const unsigned short* g_src) {
    __builtin_amdgcn_global_load_async_to_lds_b128(
        (AS1 v4i*)(void*)g_src, (AS3 v4i*)(void*)lds_dst, 0, 0);
}
#endif

union Frag {
    v16bf    v;
    unsigned u[8];
    uint4    q[2];     // two 16B quads -> ds_load_b128 / global_load_b128
};

__device__ __forceinline__ unsigned short f2bf(float f) {
    unsigned u = __builtin_bit_cast(unsigned, f);
    unsigned r = (u + 0x7FFFu + ((u >> 16) & 1u)) >> 16;   // RNE
    return (unsigned short)r;
}
__device__ __forceinline__ float bf2f(unsigned short h) {
    unsigned u = ((unsigned)h) << 16;
    return __builtin_bit_cast(float, u);
}
__device__ __forceinline__ f32x8 zero8() {
    f32x8 r;
#pragma unroll
    for (int i = 0; i < 8; i++) r[i] = 0.0f;
    return r;
}
__device__ __forceinline__ f32x8 wmma_bf(const Frag& a, const Frag& b, f32x8 c) {
    return __builtin_amdgcn_wmma_f32_16x16x32_bf16(false, a.v, false, b.v,
                                                   (short)0, c, false, false);
}

// ---------------- fp32 -> bf16 convert ----------------
__global__ void k_convert(const float* __restrict__ src,
                          unsigned short* __restrict__ dst, long n) {
    long i = (long)blockIdx.x * 256 + threadIdx.x;
    if (i < n) dst[i] = f2bf(src[i]);
}

// ---------------- conv1: 3 -> 32, scalar (1.2 GF) ----------------
__global__ void k_conv1(const float* __restrict__ x,
                        const float* __restrict__ w,
                        const float* __restrict__ bias,
                        unsigned short* __restrict__ y1) {
    long i = (long)blockIdx.x * 256 + threadIdx.x;     // SAMP*C1*81
    if (i >= (long)SAMP * C1 * P81) return;
    int p  = (int)(i % P81);
    int t  = (int)(i / P81);
    int co = t % C1;
    int s  = t / C1;
    int oy = p / 9, ox = p % 9;
    const float* xs = x + (long)s * 3 * P81;
    const float* wk = w + co * 27;
    float acc = bias[co];
#pragma unroll
    for (int ci = 0; ci < 3; ci++)
#pragma unroll
        for (int ky = 0; ky < 3; ky++) {
            int yy = oy + ky - 1;
            if ((unsigned)yy < 9u)
#pragma unroll
                for (int kx = 0; kx < 3; kx++) {
                    int xx = ox + kx - 1;
                    if ((unsigned)xx < 9u)
                        acc += xs[ci * P81 + yy * 9 + xx] * wk[ci * 9 + ky * 3 + kx];
                }
        }
    y1[i] = f2bf(acc);
}

// ---------------- BN batch stats -> (scale, shift) ----------------
__global__ void k_stats(const unsigned short* __restrict__ y, int C,
                        const float* __restrict__ gamma,
                        const float* __restrict__ beta,
                        float* __restrict__ scale, float* __restrict__ shift) {
    int c = blockIdx.x;
    __shared__ float ssum[256], ssq[256];
    float s = 0.f, q = 0.f;
    const int cnt = SAMP * P81;                       // 663552 per channel
    for (int i = threadIdx.x; i < cnt; i += 256) {
        int smp = i / P81, p = i % P81;
        float v = bf2f(y[((long)smp * C + c) * P81 + p]);
        s += v; q += v * v;
    }
    ssum[threadIdx.x] = s; ssq[threadIdx.x] = q;
    __syncthreads();
    for (int o = 128; o > 0; o >>= 1) {
        if (threadIdx.x < o) {
            ssum[threadIdx.x] += ssum[threadIdx.x + o];
            ssq[threadIdx.x]  += ssq[threadIdx.x + o];
        }
        __syncthreads();
    }
    if (threadIdx.x == 0) {
        float mu  = ssum[0] / (float)cnt;
        float var = ssq[0] / (float)cnt - mu * mu;
        float sc  = gamma[c] * rsqrtf(var + 1e-5f);
        scale[c] = sc;
        shift[c] = beta[c] - mu * sc;
    }
}

// ---------------- conv2/conv3 implicit-GEMM with WMMA (templated) ----------------
// act: [SAMP][CIN][81] bf16 RAW (pre-BN of previous layer); BN+ReLU fused on stage
// wbf: [COUT][CIN*9] bf16; out: [SAMP][COUT][81] bf16 raw (pre-BN)
template <int CIN, int COUT, int SPB>
__global__ void k_conv_wmma(const unsigned short* __restrict__ act,
                            const unsigned short* __restrict__ wbf,
                            const float* __restrict__ scale,
                            const float* __restrict__ shift,
                            const float* __restrict__ bias,
                            unsigned short* __restrict__ out) {
    extern __shared__ unsigned short lds[];
    constexpr int KK = CIN * 9;
    unsigned short* Wl  = lds;              // COUT*KK
    unsigned short* inp = lds + COUT * KK;  // CIN*121 padded image (post BN+ReLU)
    float* scl = (float*)(inp + CIN * 121); // CIN scale
    float* shf = scl + CIN;                 // CIN shift
    int tid = threadIdx.x;

#if HAVE_ASYNC_LDS
    // bulk async stage of the weight matrix (16B granules, ASYNCcnt-tracked)
    constexpr int NB = (COUT * KK) / 8;
    for (int i = tid; i < NB; i += 256)
        async_ld16(Wl + i * 8, wbf + i * 8);
    __builtin_amdgcn_s_wait_asynccnt(0);
#else
    for (int i = tid; i < COUT * KK; i += 256) Wl[i] = wbf[i];
#endif
    for (int i = tid; i < CIN; i += 256) { scl[i] = scale[i]; shf[i] = shift[i]; }

    int wave = tid >> 5, lane = tid & 31;
    int half = lane >> 4, lm = lane & 15;
    constexpr int NWT = COUT >> 4;          // n tiles (4 or 8)
    constexpr int TILES = 6 * NWT;          // m tiles: 81 -> 6x16

    int s0 = blockIdx.x * SPB;
    for (int s = s0; s < s0 + SPB; s++) {
        __syncthreads();
        const unsigned short* a = act + (long)s * CIN * P81;
        for (int i = tid; i < CIN * 121; i += 256) {
            int c = i / 121, r = i % 121, yy = r / 11, xx = r % 11;
            unsigned short v = 0;
            if (yy >= 1 && yy <= 9 && xx >= 1 && xx <= 9) {
                float f = bf2f(a[c * P81 + (yy - 1) * 9 + (xx - 1)]) * scl[c] + shf[c];
                v = f2bf(f > 0.f ? f : 0.f);          // fused BN + ReLU
            }
            inp[i] = v;
        }
        __syncthreads();
        for (int t = wave; t < TILES; t += 8) {
            int mt = t / NWT, nt = t % NWT;
            int m0 = mt << 4, n0 = nt << 4;
            int p = m0 + lm;
            int oy, ox;
            if (p > 80) { oy = 8; ox = 8; } else { oy = p / 9; ox = p % 9; }
            int base = oy * 11 + ox;                 // all patch offsets constant vs this
            f32x8 acc = zero8();
            const unsigned* wrow = (const unsigned*)(Wl + (n0 + lm) * KK);
#pragma unroll
            for (int kk = 0; kk < KK; kk += 32) {
                Frag A, Bf;
#pragma unroll
                for (int v = 0; v < 8; v++) {
                    int k0 = kk + ((v >> 2) << 4) + (half << 3) + ((v & 3) << 1);
                    int c0 = k0 / 9, r0 = k0 % 9;              // folds: kk,v are constants
                    unsigned e0 = inp[base + c0 * 121 + (r0 / 3) * 11 + (r0 % 3)];
                    int k1 = k0 + 1;
                    int c1 = k1 / 9, r1 = k1 % 9;
                    unsigned e1 = inp[base + c1 * 121 + (r1 / 3) * 11 + (r1 % 3)];
                    A.u[v] = e0 | (e1 << 16);
                }
                const uint4* qb = (const uint4*)(wrow + (kk >> 1) + (half << 2));
                Bf.q[0] = qb[0];                      // ds_load_b128 x2
                Bf.q[1] = qb[2];
                acc = wmma_bf(A, Bf, acc);
            }
            int n = n0 + lm;
            float bv = bias[n];
#pragma unroll
            for (int j = 0; j < 8; j++) {
                int m = m0 + j + 8 * half;
                if (m < 81)
                    out[((long)s * COUT + n) * P81 + m] = f2bf(acc[j] + bv);
            }
        }
    }
}

// ---------------- encoder GEMM: [8192 x 10368] x [10368 x 128] ----------------
// A = BN3+ReLU(y3 raw) fused inline; double-buffered async B staging in LDS
__global__ void k_enc(const unsigned short* __restrict__ y3,
                      const unsigned short* __restrict__ encw,  // [128][10368] bf16 == B^T
                      const float* __restrict__ scale,
                      const float* __restrict__ shift,
                      const float* __restrict__ encb,
                      float* __restrict__ e) {
    extern __shared__ unsigned short lds[];
    unsigned* Bl = (unsigned*)lds;          // 2 x [128][16] dwords (2 x 8KB)
    float* scl = (float*)(lds + 8192);      // 128 scale
    float* shf = scl + 128;                 // 128 shift
    int tid = threadIdx.x, wave = tid >> 5, lane = tid & 31;
    int half = lane >> 4, lm = lane & 15;
    int mBase = blockIdx.x * 128;

    for (int i = tid; i < 128; i += 256) { scl[i] = scale[i]; shf[i] = shift[i]; }

    f32x8 acc[8];
#pragma unroll
    for (int i = 0; i < 8; i++) acc[i] = zero8();

    const unsigned* rowA = (const unsigned*)(y3 + (long)(mBase + wave * 16 + lm) * FLATK);

    // bofs: dword offset of the active B buffer (0 or 2048)
    auto computeSlice = [&](int bofs, int kk) {
        union { uint4 q; unsigned w[4]; } ra0, ra1;           // global_load_b128 x2
        ra0.q = *(const uint4*)(rowA + (kk >> 1) + (half << 2));
        ra1.q = *(const uint4*)(rowA + (kk >> 1) + 8 + (half << 2));
        Frag A;
#pragma unroll
        for (int v = 0; v < 8; v++) {
            unsigned raw = (v < 4) ? ra0.w[v & 3] : ra1.w[v & 3];
            int k0 = kk + ((v >> 2) << 4) + (half << 3) + ((v & 3) << 1);
            int c0 = k0 / 81, c1 = (k0 + 1) / 81;
            float f0 = bf2f((unsigned short)(raw & 0xFFFFu)) * scl[c0] + shf[c0];
            float f1 = bf2f((unsigned short)(raw >> 16))     * scl[c1] + shf[c1];
            f0 = f0 > 0.f ? f0 : 0.f;                  // fused BN3 + ReLU
            f1 = f1 > 0.f ? f1 : 0.f;
            A.u[v] = (unsigned)f2bf(f0) | ((unsigned)f2bf(f1) << 16);
        }
        Frag B[8];
#pragma unroll
        for (int nt = 0; nt < 8; nt++) {
            const uint4* qp = (const uint4*)(Bl + bofs + ((nt << 4) + lm) * 16 + (half << 2));
            B[nt].q[0] = qp[0];                        // ds_load_b128 x2
            B[nt].q[1] = qp[2];
        }
#pragma unroll
        for (int nt = 0; nt < 8; nt++)
            acc[nt] = wmma_bf(A, B[nt], acc[nt]);      // back-to-back WMMA chain
    };

#if HAVE_ASYNC_LDS
    // double-buffered async pipeline: fetch slice k+1 while WMMAs consume slice k
    auto issue = [&](int ofsU16, int kk) {
#pragma unroll
        for (int i = tid; i < 512; i += 256) {   // 512 x b128 = 8KB slice
            const unsigned short* g = encw + (long)(i >> 2) * FLATK + kk + ((i & 3) << 3);
            async_ld16(lds + ofsU16 + (i << 3), g);
        }
    };
    issue(0, 0);
    int cur = 0;
    for (int kk = 0; kk < FLATK; kk += 32) {
        if (kk + 32 < FLATK) {
            issue((cur ^ 1) * 4096, kk + 32);
            __builtin_amdgcn_s_wait_asynccnt(2);   // wait for current slice only
        } else {
            __builtin_amdgcn_s_wait_asynccnt(0);
        }
        __syncthreads();
        computeSlice(cur * 2048, kk);
        __syncthreads();                            // buffer reuse fence
        cur ^= 1;
    }
#else
    for (int kk = 0; kk < FLATK; kk += 32) {
        __syncthreads();
        for (int i = tid; i < 2048; i += 256)
            Bl[i] = ((const unsigned*)(encw + (long)(i >> 4) * FLATK + kk))[i & 15];
        __syncthreads();
        computeSlice(0, kk);
    }
#endif

    int m0 = mBase + wave * 16;
#pragma unroll
    for (int nt = 0; nt < 8; nt++) {
        int n = (nt << 4) + lm;
        float bv = encb[n];
#pragma unroll
        for (int j = 0; j < 8; j++)
            e[(long)(m0 + j + 8 * half) * ENC + n] = acc[nt][j] + bv;
    }
}

// ---------------- graph filter (2 layers, K=3 taps) + action head ----------------
// one block per batch; Y matrices kept node-major [node][feature] in LDS
__global__ void k_graph(const float* __restrict__ e,
                        const float* __restrict__ gso,
                        const unsigned short* __restrict__ g1w, const float* __restrict__ g1b,
                        const unsigned short* __restrict__ g2w, const float* __restrict__ g2b,
                        const float* __restrict__ actw, const float* __restrict__ actb,
                        float* __restrict__ out) {
    extern __shared__ unsigned short lds[];
    unsigned short* St = lds;             // [n][k] = S[k][n]  (S^T row-major)
    unsigned short* Y0 = lds + 16384;     // [node][feat]
    unsigned short* Y1 = Y0 + 16384;
    int b = blockIdx.x;
    int tid = threadIdx.x, wave = tid >> 5, lane = tid & 31;
    int half = lane >> 4, lm = lane & 15;

    const float* S  = gso + (long)b * NODES * NODES;
    const float* eb = e + (long)b * NODES * ENC;
    for (int i = tid; i < 16384; i += 256) {
        int r = i >> 7, c = i & 127;
        St[i] = f2bf(S[c * NODES + r]);
        Y0[i] = f2bf(eb[i]);
    }
    __syncthreads();

    for (int layer = 0; layer < 2; layer++) {
        const unsigned short* W = layer ? g2w : g1w;     // [3][128][128] bf16
        const float* bias = layer ? g2b : g1b;
        unsigned short* Ycur = Y0;
        unsigned short* Ynxt = Y1;
        f32x8 acc[8];
#pragma unroll
        for (int i = 0; i < 8; i++) acc[i] = zero8();

        for (int k = 0; k < 3; k++) {
            // phase A: Out^T[n][g] += Y_k[n][:] . W_k[g][:]
            int m0 = wave << 4;
            const unsigned* rowA = (const unsigned*)(Ycur + (m0 + lm) * 128);
#pragma unroll
            for (int kk = 0; kk < 128; kk += 32) {
                Frag A;
                {
                    const uint4* qa = (const uint4*)(rowA + (kk >> 1) + (half << 2));
                    A.q[0] = qa[0];                    // ds_load_b128 x2
                    A.q[1] = qa[2];
                }
                Frag B[8];
#pragma unroll
                for (int nt = 0; nt < 8; nt++) {
                    const unsigned* rowB =
                        (const unsigned*)(W + ((long)(k * 128 + (nt << 4) + lm)) * 128);
                    const uint4* qb = (const uint4*)(rowB + (kk >> 1) + (half << 2));
                    B[nt].q[0] = qb[0];                // global_load_b128 x2
                    B[nt].q[1] = qb[2];
                }
#pragma unroll
                for (int nt = 0; nt < 8; nt++)
                    acc[nt] = wmma_bf(A, B[nt], acc[nt]);
            }
            // phase B: Y_{k+1} = S^T x Y_k
            if (k < 2) {
                for (int t = wave; t < 64; t += 8) {
                    int m0b = (t >> 3) << 4, n0b = (t & 7) << 4;
                    f32x8 z = zero8();
                    const unsigned* rowS = (const unsigned*)(St + (m0b + lm) * 128);
                    int col = n0b + lm;
#pragma unroll
                    for (int kk = 0; kk < 128; kk += 32) {
                        Frag A, Bf;
                        {
                            const uint4* qa = (const uint4*)(rowS + (kk >> 1) + (half << 2));
                            A.q[0] = qa[0];
                            A.q[1] = qa[2];
                        }
#pragma unroll
                        for (int v = 0; v < 8; v++) {
                            int k0 = kk + ((v >> 2) << 4) + (half << 3) + ((v & 3) << 1);
                            unsigned lo = Ycur[k0 * 128 + col];
                            unsigned hi = Ycur[(k0 + 1) * 128 + col];
                            Bf.u[v] = lo | (hi << 16);
                        }
                        z = wmma_bf(A, Bf, z);
                    }
#pragma unroll
                    for (int j = 0; j < 8; j++)
                        Ynxt[(m0b + j + 8 * half) * 128 + col] = f2bf(z[j]);
                }
            }
            __syncthreads();
            unsigned short* tmp = Ycur; Ycur = Ynxt; Ynxt = tmp;
        }
        // finalize layer: ReLU(acc + bias) -> Y0 (input of next layer / head)
#pragma unroll
        for (int nt = 0; nt < 8; nt++) {
            int g = (nt << 4) + lm;
            float bv = bias[g];
#pragma unroll
            for (int j = 0; j < 8; j++) {
                float r = acc[nt][j] + bv;
                if (r < 0.f) r = 0.f;
                Y0[((wave << 4) + j + 8 * half) * 128 + g] = f2bf(r);
            }
        }
        __syncthreads();
    }
    // action head: logits[n][a] = Y0[n][:] . actw[a][:] + actb[a]
    for (int i = tid; i < NODES * ACTS; i += 256) {
        int n = i / ACTS, a = i % ACTS;
        float s = actb[a];
        for (int g = 0; g < 128; g++)
            s += bf2f(Y0[n * 128 + g]) * actw[a * 128 + g];
        out[((long)b * NODES + n) * ACTS + a] = s;
    }
}

// ---------------- host launch ----------------
extern "C" void kernel_launch(void* const* d_in, const int* in_sizes, int n_in,
                              void* d_out, int out_size, void* d_ws, size_t ws_size,
                              hipStream_t stream) {
    const float* states = (const float*)d_in[0];
    const float* gso    = (const float*)d_in[1];
    const float* c1_w = (const float*)d_in[2];
    const float* c1_b = (const float*)d_in[3];
    const float* c1_g = (const float*)d_in[4];
    const float* c1_be= (const float*)d_in[5];
    const float* c2_w = (const float*)d_in[6];
    const float* c2_b = (const float*)d_in[7];
    const float* c2_g = (const float*)d_in[8];
    const float* c2_be= (const float*)d_in[9];
    const float* c3_w = (const float*)d_in[10];
    const float* c3_b = (const float*)d_in[11];
    const float* c3_g = (const float*)d_in[12];
    const float* c3_be= (const float*)d_in[13];
    const float* enc_w= (const float*)d_in[14];
    const float* enc_b= (const float*)d_in[15];
    const float* g1_w = (const float*)d_in[16];
    const float* g1_b = (const float*)d_in[17];
    const float* g2_w = (const float*)d_in[18];
    const float* g2_b = (const float*)d_in[19];
    const float* act_w= (const float*)d_in[20];
    const float* act_b= (const float*)d_in[21];
    float* out = (float*)d_out;

    char* ws = (char*)d_ws;
    size_t off = 0;
    auto carve = [&](size_t bytes) -> char* {
        char* p = ws + off;
        off += (bytes + 255) & ~(size_t)255;
        return p;
    };
    const long n_w2  = 64L * 32 * 9;           // 18432
    const long n_w3  = 128L * 64 * 9;          // 73728
    const long n_enc = 128L * FLATK;           // 1327104
    const long n_g   = 3L * 128 * 128;         // 49152
    const long n_y1  = (long)SAMP * C1 * P81;  // 21.2M
    const long n_y2  = (long)SAMP * C2 * P81;  // 42.5M
    const long n_y3  = (long)SAMP * C3 * P81;  // 84.9M

    unsigned short* w2bf  = (unsigned short*)carve(n_w2 * 2);
    unsigned short* w3bf  = (unsigned short*)carve(n_w3 * 2);
    unsigned short* encbf = (unsigned short*)carve(n_enc * 2);
    unsigned short* g1bf  = (unsigned short*)carve(n_g * 2);
    unsigned short* g2bf  = (unsigned short*)carve(n_g * 2);
    float* sc1 = (float*)carve(32 * 4);  float* sh1 = (float*)carve(32 * 4);
    float* sc2 = (float*)carve(64 * 4);  float* sh2 = (float*)carve(64 * 4);
    float* sc3 = (float*)carve(128 * 4); float* sh3 = (float*)carve(128 * 4);
    unsigned short* y1 = (unsigned short*)carve(n_y1 * 2);
    unsigned short* y2 = (unsigned short*)carve(n_y2 * 2);
    unsigned short* y3 = (unsigned short*)carve(n_y3 * 2);
    float* e = (float*)carve((long)SAMP * ENC * 4);

    auto blocks = [](long n) { return (unsigned)((n + 255) / 256); };

    // weights -> bf16
    k_convert<<<blocks(n_w2),  256, 0, stream>>>(c2_w,  w2bf,  n_w2);
    k_convert<<<blocks(n_w3),  256, 0, stream>>>(c3_w,  w3bf,  n_w3);
    k_convert<<<blocks(n_enc), 256, 0, stream>>>(enc_w, encbf, n_enc);
    k_convert<<<blocks(n_g),   256, 0, stream>>>(g1_w,  g1bf,  n_g);
    k_convert<<<blocks(n_g),   256, 0, stream>>>(g2_w,  g2bf,  n_g);

    // conv1 (raw) + BN1 stats
    k_conv1<<<blocks(n_y1), 256, 0, stream>>>(states, c1_w, c1_b, y1);
    k_stats<<<C1, 256, 0, stream>>>(y1, C1, c1_g, c1_be, sc1, sh1);

    // conv2 (implicit GEMM WMMA; BN1+ReLU fused on input stage) + BN2 stats
    size_t lds2 = (size_t)(C2 * C1 * 9 + C1 * 121) * 2 + (size_t)C1 * 8;   // 44,864 B
    k_conv_wmma<C1, C2, 16><<<SAMP / 16, 256, lds2, stream>>>(y1, w2bf, sc1, sh1, c2_b, y2);
    k_stats<<<C2, 256, 0, stream>>>(y2, C2, c2_g, c2_be, sc2, sh2);

    // conv3 (implicit GEMM WMMA; BN2+ReLU fused) + BN3 stats
    size_t lds3 = (size_t)(C3 * C2 * 9 + C2 * 121) * 2 + (size_t)C2 * 8;   // 163,456 B
    k_conv_wmma<C2, C3, 16><<<SAMP / 16, 256, lds3, stream>>>(y2, w3bf, sc2, sh2, c3_b, y3);
    k_stats<<<C3, 256, 0, stream>>>(y3, C3, c3_g, c3_be, sc3, sh3);

    // encoder GEMM (WMMA; BN3+ReLU fused on A; async double-buffered B staging)
    k_enc<<<SAMP / 128, 256, 2 * 8192 + 1024, stream>>>(y3, encbf, sc3, sh3, enc_b, e);

    // graph filters + head (WMMA in LDS), one block per batch
    k_graph<<<BATCH, 256, 3 * 16384 * 2, stream>>>(e, gso, g1bf, g1_b,
                                                   g2bf, g2_b, act_w, act_b, out);
}
